// RGCN_78005196030468
// MI455X (gfx1250) — matrix operand order repaired and checked by reading, not compiled, not run on hardware
//
#include <hip/hip_runtime.h>

#define D 128

typedef __bf16 bf16_t;
typedef bf16_t bf16x16 __attribute__((ext_vector_type(16)));
typedef float  f32x8   __attribute__((ext_vector_type(8)));

__device__ __forceinline__ bf16_t f2bf(float f) {
    unsigned u = __float_as_uint(f);
    unsigned r = (u + 0x7FFFu + ((u >> 16) & 1u)) >> 16;   // round-to-nearest-even
    unsigned short s = (unsigned short)r;
    return __builtin_bit_cast(bf16_t, s);
}
__device__ __forceinline__ float bf2f(bf16_t h) {
    unsigned short s = __builtin_bit_cast(unsigned short, h);
    return __uint_as_float(((unsigned)s) << 16);
}

// Native device-scope f32 atomic add (guaranteed global_atomic_add_f32, no CAS loop).
__device__ __forceinline__ void atomic_add_f32(float* p, float v) {
    asm volatile("global_atomic_add_f32 %0, %1, off scope:SCOPE_DEV"
                 :: "v"(p), "v"(v) : "memory");
}

// -------------------------------------------------- weight split + frag-swizzle
// Emits bf16 hi/lo weights pre-packed in WMMA B-fragment order so each lane's
// 16-element fragment is 32 contiguous bytes:
//   packed[((ks*128 + n)*2 + khalf)*16 + e] = W[(ks*32 + khalf*16 + e)*128 + n]
__global__ __launch_bounds__(256) void k_split(const float* __restrict__ w,
                                               bf16_t* __restrict__ hi,
                                               bf16_t* __restrict__ lo, int n) {
    int i = blockIdx.x * blockDim.x + threadIdx.x;   // packed index
    if (i >= n) return;
    int e     = i & 15;
    int khalf = (i >> 4) & 1;
    int col   = (i >> 5) & 127;
    int ks    = (i >> 12) & 3;
    int k     = ks * 32 + khalf * 16 + e;
    float f = w[(size_t)k * D + col];
    bf16_t h = f2bf(f);
    hi[i] = h;
    lo[i] = f2bf(f - bf2f(h));
}

// ------------------------------------------------------- dual GEMM (bf16x3 WMMA)
// Out = A[N,128] @ W[128,128] for W_rel and W_root simultaneously.
// One wave computes a 16-row strip (16x128 of both outputs).
// NOTE: A and outRel may alias (in-place for layer 2): every A element this wave
// needs is loaded into registers before any store to the same rows.
__global__ __launch_bounds__(128) void k_gemm_dual(
        const float* A,
        const bf16_t* __restrict__ relHi, const bf16_t* __restrict__ relLo,
        const bf16_t* __restrict__ rootHi, const bf16_t* __restrict__ rootLo,
        float* outRel, float* outRoot, int nTiles) {
    int wave = (blockIdx.x * blockDim.x + threadIdx.x) >> 5;
    if (wave >= nTiles) return;
    int lane  = threadIdx.x & 31;
    int mlane = lane & 15;
    int khalf = lane >> 4;
    int m0 = wave * 16;

    // ---- A fragments (16x32 bf16 layout), hi + lo split, all 4 k-steps ----
    // Per k-step each lane needs one contiguous 16-float run:
    //   k = ks*32 + khalf*16 + {0..15}
    bf16x16 ahi[4], alo[4];
    const float* arow = A + (size_t)(m0 + mlane) * D;
#pragma unroll
    for (int ks = 0; ks < 4; ++ks) {
        const float* p = arow + ks * 32 + khalf * 16;
        float4 q0 = *reinterpret_cast<const float4*>(p);
        float4 q1 = *reinterpret_cast<const float4*>(p + 4);
        float4 q2 = *reinterpret_cast<const float4*>(p + 8);
        float4 q3 = *reinterpret_cast<const float4*>(p + 12);
        float vals[16] = {q0.x, q0.y, q0.z, q0.w, q1.x, q1.y, q1.z, q1.w,
                          q2.x, q2.y, q2.z, q2.w, q3.x, q3.y, q3.z, q3.w};
#pragma unroll
        for (int e = 0; e < 16; ++e) {
            float v = vals[e];
            bf16_t h = f2bf(v);
            ahi[ks][e] = h;
            alo[ks][e] = f2bf(v - bf2f(h));
        }
    }

    for (int wsel = 0; wsel < 2; ++wsel) {
        const bf16_t* whi = wsel ? rootHi : relHi;
        const bf16_t* wlo = wsel ? rootLo : relLo;
        float* out        = wsel ? outRoot : outRel;
#pragma unroll
        for (int nt = 0; nt < 8; ++nt) {
            int n0 = nt * 16;
            f32x8 acc = {0.f, 0.f, 0.f, 0.f, 0.f, 0.f, 0.f, 0.f};
#pragma unroll
            for (int ks = 0; ks < 4; ++ks) {
                // ---- B fragment: 32 contiguous bytes in pre-swizzled layout
                size_t foff = (size_t)(((ks * 128 + n0 + mlane) * 2 + khalf)) * 16;
                bf16x16 bhi = *reinterpret_cast<const bf16x16*>(whi + foff);
                bf16x16 blo = *reinterpret_cast<const bf16x16*>(wlo + foff);
                acc = __builtin_amdgcn_wmma_f32_16x16x32_bf16(
                        false, ahi[ks], false, bhi, (short)0, acc, false, false);
                acc = __builtin_amdgcn_wmma_f32_16x16x32_bf16(
                        false, ahi[ks], false, blo, (short)0, acc, false, false);
                acc = __builtin_amdgcn_wmma_f32_16x16x32_bf16(
                        false, alo[ks], false, bhi, (short)0, acc, false, false);
            }
            // ---- store C (f32 16x16 layout): VGPR r -> row m0+r(+8), col = lane&15
#pragma unroll
            for (int r = 0; r < 8; ++r) {
                int row = m0 + r + khalf * 8;
                out[(size_t)row * D + n0 + mlane] = acc[r];
            }
        }
    }
}

// ---------------------------------------------------------------- zero scratch
__global__ __launch_bounds__(256) void k_zero(float* __restrict__ p, int n) {
    int i = blockIdx.x * blockDim.x + threadIdx.x;
    if (i < n) p[i] = 0.f;
}

// ------------------------------------------------------------- edge scatter-add
// One wave per edge: gather msg[src] (float4 per lane), atomic-add into agg[dst].
__global__ __launch_bounds__(256) void k_scatter(const float* __restrict__ msg,
                                                 const int* __restrict__ src,
                                                 const int* __restrict__ dst,
                                                 float* __restrict__ agg,
                                                 float* __restrict__ cnt, int E) {
    int edge = blockIdx.x * (blockDim.x >> 5) + (threadIdx.x >> 5);
    if (edge >= E) return;
    int lane = threadIdx.x & 31;
    int s = src[edge];
    int d = dst[edge];
    const float4 v = *reinterpret_cast<const float4*>(msg + (size_t)s * D + lane * 4);
    float* base = agg + (size_t)d * D + lane * 4;
    atomic_add_f32(base + 0, v.x);
    atomic_add_f32(base + 1, v.y);
    atomic_add_f32(base + 2, v.z);
    atomic_add_f32(base + 3, v.w);
    if (lane == 0) atomic_add_f32(cnt + d, 1.0f);
}

// ------------------------------------------------------------------- finalize
// out = agg/max(cnt,1) + root + bias, optional ReLU. One thread per 4 floats.
__global__ __launch_bounds__(256) void k_finalize(const float* __restrict__ agg,
                                                  const float* __restrict__ cnt,
                                                  const float* __restrict__ root,
                                                  const float* __restrict__ bias,
                                                  float* __restrict__ out,
                                                  int N, int relu) {
    int i = blockIdx.x * blockDim.x + threadIdx.x;
    if (i >= N * (D / 4)) return;
    int row = i >> 5;            // D/4 == 32 chunks per row
    int c4  = (i & 31) * 4;
    float inv = 1.0f / fmaxf(cnt[row], 1.0f);
    float4 a = *reinterpret_cast<const float4*>(agg  + (size_t)row * D + c4);
    float4 r = *reinterpret_cast<const float4*>(root + (size_t)row * D + c4);
    float4 b = *reinterpret_cast<const float4*>(bias + c4);
    float4 o;
    o.x = fmaf(a.x, inv, r.x + b.x);
    o.y = fmaf(a.y, inv, r.y + b.y);
    o.z = fmaf(a.z, inv, r.z + b.z);
    o.w = fmaf(a.w, inv, r.w + b.w);
    if (relu) {
        o.x = fmaxf(o.x, 0.f); o.y = fmaxf(o.y, 0.f);
        o.z = fmaxf(o.z, 0.f); o.w = fmaxf(o.w, 0.f);
    }
    *reinterpret_cast<float4*>(out + (size_t)row * D + c4) = o;
}

extern "C" void kernel_launch(void* const* d_in, const int* in_sizes, int n_in,
                              void* d_out, int out_size, void* d_ws, size_t ws_size,
                              hipStream_t stream) {
    const float* x      = (const float*)d_in[0];
    const int*   ei     = (const int*)d_in[1];
    const float* W1_rel  = (const float*)d_in[2];
    const float* W1_root = (const float*)d_in[3];
    const float* b1      = (const float*)d_in[4];
    const float* W2_rel  = (const float*)d_in[5];
    const float* W2_root = (const float*)d_in[6];
    const float* b2      = (const float*)d_in[7];
    float* out = (float*)d_out;

    const int N = in_sizes[0] / D;      // 50000
    const int E = in_sizes[1] / 2;      // 800000
    const int* src = ei;
    const int* dst = ei + E;

    // ---- workspace layout (fp32 counts) ----
    float* XC  = (float*)d_ws;                 // rel-transformed features / H
    float* XR  = XC  + (size_t)N * D;          // root-transformed features
    float* AGG = XR  + (size_t)N * D;          // scatter accumulation
    float* CNT = AGG + (size_t)N * D;          // degree counts (contiguous w/ AGG)
    bf16_t* wp = (bf16_t*)(CNT + N);
    const int WN = D * D;
    bf16_t* w1rel_hi  = wp;            bf16_t* w1rel_lo  = wp + 1 * WN;
    bf16_t* w1root_hi = wp + 2 * WN;   bf16_t* w1root_lo = wp + 3 * WN;
    bf16_t* w2rel_hi  = wp + 4 * WN;   bf16_t* w2rel_lo  = wp + 5 * WN;
    bf16_t* w2root_hi = wp + 6 * WN;   bf16_t* w2root_lo = wp + 7 * WN;

    const int splitBlocks = (WN + 255) / 256;
    k_split<<<splitBlocks, 256, 0, stream>>>(W1_rel,  w1rel_hi,  w1rel_lo,  WN);
    k_split<<<splitBlocks, 256, 0, stream>>>(W1_root, w1root_hi, w1root_lo, WN);
    k_split<<<splitBlocks, 256, 0, stream>>>(W2_rel,  w2rel_hi,  w2rel_lo,  WN);
    k_split<<<splitBlocks, 256, 0, stream>>>(W2_root, w2root_hi, w2root_lo, WN);

    const int nTiles     = N / 16;                    // 3125
    const int gemmBlocks = (nTiles + 3) / 4;          // 4 waves / 128-thread block
    const int zc         = N * (D + 1);               // AGG + CNT contiguous
    const int zeroBlocks = (zc + 255) / 256;
    const int scatBlocks = (E + 7) / 8;               // 8 edges (waves) per block
    const int finBlocks  = (N * 32 + 255) / 256;

    // ---------------- layer 1 ----------------
    k_gemm_dual<<<gemmBlocks, 128, 0, stream>>>(x, w1rel_hi, w1rel_lo,
                                                w1root_hi, w1root_lo,
                                                XC, XR, nTiles);
    k_zero<<<zeroBlocks, 256, 0, stream>>>(AGG, zc);
    k_scatter<<<scatBlocks, 256, 0, stream>>>(XC, src, dst, AGG, CNT, E);
    k_finalize<<<finBlocks, 256, 0, stream>>>(AGG, CNT, XR, b1, XC, N, 1); // H -> XC

    // ---------------- layer 2 ----------------
    k_gemm_dual<<<gemmBlocks, 128, 0, stream>>>(XC, w2rel_hi, w2rel_lo,
                                                w2root_hi, w2root_lo,
                                                XC /*in-place safe*/, XR, nTiles);
    k_zero<<<zeroBlocks, 256, 0, stream>>>(AGG, zc);
    k_scatter<<<scatBlocks, 256, 0, stream>>>(XC, src, dst, AGG, CNT, E);
    k_finalize<<<finBlocks, 256, 0, stream>>>(AGG, CNT, XR, b2, out, N, 0);
}
